// MultiHeadedAttention_32203664785461
// MI455X (gfx1250) — matrix operand order repaired and checked by reading, not compiled, not run on hardware
//
#include <hip/hip_runtime.h>
#include <hip/hip_bf16.h>
#include <math.h>

typedef __attribute__((ext_vector_type(16))) _Float16 v16h;
typedef __attribute__((ext_vector_type(8)))  float    v8f;

#define S_LEN 4096
#define HID_N 1024
#define HQ_N  16
#define HKV_N 4
#define DH    64
#define KVW   256   // HKV*DH

static __device__ __forceinline__ v8f wmma_f16(const v16h& a, const v16h& b, const v8f& c) {
  // D = A(16x32 f16) * B(32x16 f16) + C(16x16 f32)
  return __builtin_amdgcn_wmma_f32_16x16x32_f16(false, a, false, b, (short)0, c, false, false);
}

// One 16-byte-per-lane async global->LDS copy (512B per wave-issue).
// lds_off: per-lane LDS byte offset; g: per-lane global address.
static __device__ __forceinline__ void async_copy16(unsigned lds_off, const void* g) {
  asm volatile("global_load_async_to_lds_b128 %0, %1, off"
               :: "v"(lds_off), "v"(g) : "memory");
}

// ---------------------------------------------------------------------------
// Fragment layouts per CDNA5 ISA 7.12.2:
// A (16x32 f16): lane m=lane&15, half=lane>>4.
//   VGPR v 0..7 : K = half*16 + 2v (+1) (contiguous k order used in this file)
// B (32x16 f16): lane n=lane&15, khalf=lane>>4. VGPR v: K = khalf*16 + 2v (+1) (contiguous order chosen here, same as A)
// C/D (16x16 f32): col n=lane&15, VGPR v holds row = v + (lane>>4)*8.
// ---------------------------------------------------------------------------

// Kernel 1: fused QKV projection. One wave computes a 64x16 output tile
// (B-fragment reused across 4 A-fragments -> 4 WMMAs per strided weight load).
// grid = (64 row tiles, 96 col tiles), block = 32.
__global__ void qkv_kernel(const float* __restrict__ x,
                           const float* __restrict__ Wq,
                           const float* __restrict__ Wk,
                           const float* __restrict__ Wv,
                           float* __restrict__ Qf,
                           float* __restrict__ Kf,
                           float* __restrict__ Vf) {
  const int lane = threadIdx.x & 31;
  const int m    = lane & 15;
  const int half = lane >> 4;
  const int rt   = blockIdx.x;   // 0..63
  const int ct   = blockIdx.y;   // 0..95

  const float* W; float* Out; int N; int col0;
  if (ct < 64)      { W = Wq; Out = Qf; N = HID_N; col0 = ct * 16; }
  else if (ct < 80) { W = Wk; Out = Kf; N = KVW;   col0 = (ct - 64) * 16; }
  else              { W = Wv; Out = Vf; N = KVW;   col0 = (ct - 80) * 16; }

  const int row0 = rt * 64;
  const int n    = col0 + m;
  v8f acc[4] = {};

  for (int k0 = 0; k0 < HID_N; k0 += 32) {
    v16h b;
    const float* wr = W + (size_t)(k0 + half * 16) * N + n;
#pragma unroll
    for (int v = 0; v < 8; ++v) {
      b[2 * v]     = (_Float16)wr[(size_t)(2 * v) * N];
      b[2 * v + 1] = (_Float16)wr[(size_t)(2 * v + 1) * N];
    }
#pragma unroll
    for (int r = 0; r < 4; ++r) {
      v16h a;
      const float* xr = x + (size_t)(row0 + r * 16 + m) * HID_N + k0;
#pragma unroll
      for (int v = 0; v < 4; ++v) {
        const int kk  = half * 16 + 2 * v;
        const int kk2 = half * 16 + 8 + 2 * v;
        a[2 * v]     = (_Float16)xr[kk];
        a[2 * v + 1] = (_Float16)xr[kk + 1];
        a[2 * v + 8] = (_Float16)xr[kk2];
        a[2 * v + 9] = (_Float16)xr[kk2 + 1];
      }
      acc[r] = wmma_f16(a, b, acc[r]);
    }
  }
#pragma unroll
  for (int r = 0; r < 4; ++r)
#pragma unroll
    for (int v = 0; v < 8; ++v)
      Out[(size_t)(row0 + r * 16 + half * 8 + v) * N + col0 + m] = acc[r][v];
}

// Kernel 2: RoPE (pairs i, i+32 within each 64-wide head) + fp32 -> f16.
__global__ void rope_kernel(const float* __restrict__ In,
                            _Float16* __restrict__ Out, int H) {
  const int idx = blockIdx.x * blockDim.x + threadIdx.x;
  const int total = S_LEN * H * 32;
  if (idx >= total) return;
  const int s = idx / (H * 32);
  const int r = idx - s * (H * 32);
  const int h = r >> 5;
  const int i = r & 31;
  const float ang = (float)s * __powf(10000.0f, -(float)i / 32.0f);
  float c, sn;
  __sincosf(ang, &sn, &c);
  const size_t base = (size_t)s * H * DH + h * DH + i;
  const float t1 = In[base];
  const float t2 = In[base + 32];
  Out[base]      = (_Float16)(t1 * c - t2 * sn);
  Out[base + 32] = (_Float16)(t2 * c + t1 * sn);
}

// Kernel 2b: plain fp32 -> f16 conversion (V path, no RoPE).
__global__ void cvt_kernel(const float* __restrict__ In,
                           _Float16* __restrict__ Out, int total) {
  const int idx = blockIdx.x * blockDim.x + threadIdx.x;
  if (idx < total) Out[idx] = (_Float16)In[idx];
}

// Stage one 32x64 f16 KV-block row-tile (4 KB) into LDS: 8 async b128 issues.
static __device__ __forceinline__ void stage_tile(const _Float16* __restrict__ src,
                                                  int kv0, int kvh,
                                                  _Float16* buf,
                                                  int crow, int cbyte) {
#pragma unroll
  for (int c = 0; c < 8; ++c) {
    const int row = c * 4 + crow;                           // 0..31
    const unsigned loff = (unsigned)(size_t)buf + row * 128 + cbyte;
    const char* g = (const char*)(src + (size_t)(kv0 + row) * KVW + kvh * DH) + cbyte;
    async_copy16(loff, g);
  }
}

// Kernel 3: causal flash attention with GQA. One wave per (q-tile, head).
// K/V blocks are double-buffered in LDS via global_load_async_to_lds_b128,
// drained with s_wait_asynccnt. grid = (256, 16), block = 32.
__global__ void attn_kernel(const _Float16* __restrict__ Qh,
                            const _Float16* __restrict__ Kh,
                            const _Float16* __restrict__ Vh,
                            _Float16* __restrict__ Ctx) {
  __shared__ _Float16 ldsK[2][32 * DH];   // 2 x 4 KB
  __shared__ _Float16 ldsV[2][32 * DH];   // 2 x 4 KB
  __shared__ float    pLds[16 * 32];      // 2 KB

  const int lane = threadIdx.x & 31;
  const int m    = lane & 15;
  const int half = lane >> 4;
  const int qt   = blockIdx.x;        // 0..255
  const int h    = blockIdx.y;        // 0..15
  const int kvh  = h >> 2;            // GQA group of 4
  const int row0 = qt * 16;
  const float scale = 0.125f;         // 1/sqrt(64)

  // async-copy geometry: 8 lanes per 128B row, 4 rows per 512B issue
  const int crow  = lane >> 3;        // 0..3
  const int cbyte = (lane & 7) * 16;  // 0..112

  // Q fragments: 16x64 -> two 16x32 A-fragments (contiguous 16B chunks).
  v16h qa[2];
  const _Float16* qr = Qh + (size_t)(row0 + m) * HID_N + h * DH;
#pragma unroll
  for (int f = 0; f < 2; ++f) {
#pragma unroll
    for (int v = 0; v < 4; ++v) {
      const int kk  = f * 32 + half * 16 + 2 * v;
      const int kk2 = f * 32 + half * 16 + 8 + 2 * v;
      qa[f][2 * v]     = qr[kk];
      qa[f][2 * v + 1] = qr[kk + 1];
      qa[f][2 * v + 8] = qr[kk2];
      qa[f][2 * v + 9] = qr[kk2 + 1];
    }
  }

  float mrow[8], lrow[8];
#pragma unroll
  for (int v = 0; v < 8; ++v) { mrow[v] = -1.0e30f; lrow[v] = 0.0f; }
  v8f o[4] = {};

  const int nkb = qt / 2 + 1;  // 32-wide KV blocks covering causal range

  // prologue: stage block 0 into buffer 0 (16 async copies)
  stage_tile(Kh, 0, kvh, &ldsK[0][0], crow, cbyte);
  stage_tile(Vh, 0, kvh, &ldsV[0][0], crow, cbyte);

  for (int kb = 0; kb < nkb; ++kb) {
    const int kv0 = kb * 32;
    const int buf = kb & 1;

    if (kb + 1 < nkb) {
      // prefetch next block into the other buffer, then wait for current:
      // async loads complete in order, so <=16 outstanding => block kb resident.
      stage_tile(Kh, kv0 + 32, kvh, &ldsK[buf ^ 1][0], crow, cbyte);
      stage_tile(Vh, kv0 + 32, kvh, &ldsV[buf ^ 1][0], crow, cbyte);
      asm volatile("s_wait_asynccnt 0x10" ::: "memory");
    } else {
      asm volatile("s_wait_asynccnt 0x0" ::: "memory");
    }

    // ---- scores S = Q * K^T : two 16x16 f32 tiles (K from LDS) ----
    v8f sc[2] = {};
#pragma unroll
    for (int t = 0; t < 2; ++t) {
      const _Float16* kr = &ldsK[buf][(t * 16 + m) * DH];
#pragma unroll
      for (int f = 0; f < 2; ++f) {
        v16h b;
#pragma unroll
        for (int v = 0; v < 8; ++v) {
          const int d = f * 32 + half * 16 + 2 * v;
          b[2 * v]     = kr[d];
          b[2 * v + 1] = kr[d + 1];
        }
        sc[t] = wmma_f16(qa[f], b, sc[t]);
      }
    }

    // ---- online softmax (rows live in 16-lane halves) ----
#pragma unroll
    for (int v = 0; v < 8; ++v) {
      const int qg = row0 + half * 8 + v;          // global q row
      float s0 = sc[0][v] * scale;
      float s1 = sc[1][v] * scale;
      if (kv0 + m > qg)      s0 = -1.0e30f;        // causal mask
      if (kv0 + 16 + m > qg) s1 = -1.0e30f;
      float bm = fmaxf(s0, s1);
      bm = fmaxf(bm, __shfl_xor(bm, 1, 32));
      bm = fmaxf(bm, __shfl_xor(bm, 2, 32));
      bm = fmaxf(bm, __shfl_xor(bm, 4, 32));
      bm = fmaxf(bm, __shfl_xor(bm, 8, 32));
      const float newm  = fmaxf(mrow[v], bm);
      const float alpha = __expf(mrow[v] - newm);
      const float p0 = __expf(s0 - newm);
      const float p1 = __expf(s1 - newm);
      float rs = p0 + p1;
      rs += __shfl_xor(rs, 1, 32);
      rs += __shfl_xor(rs, 2, 32);
      rs += __shfl_xor(rs, 4, 32);
      rs += __shfl_xor(rs, 8, 32);
      lrow[v] = lrow[v] * alpha + rs;
      mrow[v] = newm;
#pragma unroll
      for (int t = 0; t < 4; ++t) o[t][v] *= alpha;
      pLds[(half * 8 + v) * 32 + m]      = p0;
      pLds[(half * 8 + v) * 32 + 16 + m] = p1;
    }
    __syncthreads();

    // ---- relayout P (C-layout in LDS) into an A-fragment ----
    v16h pa;
#pragma unroll
    for (int v = 0; v < 4; ++v) {
      const int kk  = half * 16 + 2 * v;
      const int kk2 = half * 16 + 8 + 2 * v;
      pa[2 * v]     = (_Float16)pLds[m * 32 + kk];
      pa[2 * v + 1] = (_Float16)pLds[m * 32 + kk + 1];
      pa[2 * v + 8] = (_Float16)pLds[m * 32 + kk2];
      pa[2 * v + 9] = (_Float16)pLds[m * 32 + kk2 + 1];
    }
    __syncthreads();

    // ---- O += P * V : four 16x16 d-tiles (V from LDS, K = 32 kv rows) ----
#pragma unroll
    for (int t = 0; t < 4; ++t) {
      const _Float16* vr = &ldsV[buf][half * 16 * DH + t * 16 + m];
      v16h b;
#pragma unroll
      for (int v = 0; v < 8; ++v) {
        b[2 * v]     = vr[(2 * v) * DH];
        b[2 * v + 1] = vr[(2 * v + 1) * DH];
      }
      o[t] = wmma_f16(pa, b, o[t]);
    }
  }

  // finalize: divide by row sums, store ctx in f16 (layout [S][HQ*D])
#pragma unroll
  for (int v = 0; v < 8; ++v) {
    const float inv_l = 1.0f / lrow[v];
    const size_t rbase = (size_t)(row0 + half * 8 + v) * HID_N + h * DH;
#pragma unroll
    for (int t = 0; t < 4; ++t)
      Ctx[rbase + t * 16 + m] = (_Float16)(o[t][v] * inv_l);
  }
}

// Kernel 4: out = ctx(f16) @ Wo(f32->f16). One wave per 64x16 tile.
// grid = (64, 64), block = 32.
__global__ void out_kernel(const _Float16* __restrict__ Ctx,
                           const float* __restrict__ Wo,
                           float* __restrict__ out) {
  const int lane = threadIdx.x & 31;
  const int m    = lane & 15;
  const int half = lane >> 4;
  const int row0 = blockIdx.x * 64;
  const int col0 = blockIdx.y * 16;
  const int n    = col0 + m;
  v8f acc[4] = {};

  for (int k0 = 0; k0 < HID_N; k0 += 32) {
    v16h b;
    const float* wr = Wo + (size_t)(k0 + half * 16) * HID_N + n;
#pragma unroll
    for (int v = 0; v < 8; ++v) {
      b[2 * v]     = (_Float16)wr[(size_t)(2 * v) * HID_N];
      b[2 * v + 1] = (_Float16)wr[(size_t)(2 * v + 1) * HID_N];
    }
#pragma unroll
    for (int r = 0; r < 4; ++r) {
      v16h a;
      const _Float16* ar = Ctx + (size_t)(row0 + r * 16 + m) * HID_N + k0;
#pragma unroll
      for (int v = 0; v < 4; ++v) {
        const int kk  = half * 16 + 2 * v;
        const int kk2 = half * 16 + 8 + 2 * v;
        a[2 * v]     = ar[kk];
        a[2 * v + 1] = ar[kk + 1];
        a[2 * v + 8] = ar[kk2];
        a[2 * v + 9] = ar[kk2 + 1];
      }
      acc[r] = wmma_f16(a, b, acc[r]);
    }
  }
#pragma unroll
  for (int r = 0; r < 4; ++r)
#pragma unroll
    for (int v = 0; v < 8; ++v)
      out[(size_t)(row0 + r * 16 + half * 8 + v) * HID_N + col0 + m] = acc[r][v];
}

extern "C" void kernel_launch(void* const* d_in, const int* in_sizes, int n_in,
                              void* d_out, int out_size, void* d_ws, size_t ws_size,
                              hipStream_t stream) {
  const float* x  = (const float*)d_in[0];
  const float* Wq = (const float*)d_in[1];
  const float* Wk = (const float*)d_in[2];
  const float* Wv = (const float*)d_in[3];
  const float* Wo = (const float*)d_in[4];
  float* out = (float*)d_out;

  char* ws = (char*)d_ws;
  const size_t MB = 1024u * 1024u;
  float*    Qf  = (float*)(ws + 0 * MB);     // 16 MB : S x 1024 f32
  float*    Kf  = (float*)(ws + 16 * MB);    //  4 MB : S x 256  f32
  float*    Vf  = (float*)(ws + 20 * MB);    //  4 MB : S x 256  f32
  _Float16* Qh  = (_Float16*)(ws + 24 * MB); //  8 MB : S x 1024 f16 (post-RoPE)
  _Float16* Kh  = (_Float16*)(ws + 32 * MB); //  2 MB : S x 256  f16 (post-RoPE)
  _Float16* Vh  = (_Float16*)(ws + 34 * MB); //  2 MB : S x 256  f16
  _Float16* Ctx = (_Float16*)(ws + 36 * MB); //  8 MB : S x 1024 f16

  qkv_kernel<<<dim3(S_LEN / 64, 96), 32, 0, stream>>>(x, Wq, Wk, Wv, Qf, Kf, Vf);
  rope_kernel<<<dim3((S_LEN * HQ_N * 32 + 255) / 256), 256, 0, stream>>>(Qf, Qh, HQ_N);
  rope_kernel<<<dim3((S_LEN * HKV_N * 32 + 255) / 256), 256, 0, stream>>>(Kf, Kh, HKV_N);
  cvt_kernel<<<dim3((S_LEN * KVW + 255) / 256), 256, 0, stream>>>(Vf, Vh, S_LEN * KVW);
  attn_kernel<<<dim3(S_LEN / 16, HQ_N), 32, 0, stream>>>(Qh, Kh, Vh, Ctx);
  out_kernel<<<dim3(S_LEN / 64, HID_N / 16), 32, 0, stream>>>(Ctx, Wo, out);
}
